// BertGNNGru_8985071583334
// MI455X (gfx1250) — matrix operand order repaired and checked by reading, not compiled, not run on hardware
//
#include <hip/hip_runtime.h>
#include <hip/hip_bf16.h>

// ---------------------------------------------------------------------------
// Problem dims (fixed by reference)
// ---------------------------------------------------------------------------
#define B_   1024
#define T_   512
#define D_   256
#define H_   256
#define G_   768   // 3*H

typedef __attribute__((ext_vector_type(16))) __bf16 v16bf;
typedef __attribute__((ext_vector_type(8)))  __bf16 v8bf;
typedef __attribute__((ext_vector_type(8)))  float  v8f;
typedef __attribute__((ext_vector_type(4)))  float  v4f;

typedef __attribute__((ext_vector_type(4))) unsigned int u32x4;
typedef __attribute__((ext_vector_type(4))) int          i32x4;
typedef __attribute__((ext_vector_type(8))) int          i32x8;

// fp32 -> bf16 (round to nearest even), stored as ushort
static __device__ __forceinline__ unsigned short f2bfu(float f) {
    unsigned u = __builtin_bit_cast(unsigned, f);
    u += 0x7FFFu + ((u >> 16) & 1u);
    return (unsigned short)(u >> 16);
}
static __device__ __forceinline__ float bf2f(unsigned short s) {
    unsigned u = ((unsigned)s) << 16;
    return __builtin_bit_cast(float, u);
}
static __device__ __forceinline__ v16bf mk16(v8bf lo, v8bf hi) {
    v16bf r;
#pragma unroll
    for (int i = 0; i < 8; ++i) { r[i] = lo[i]; r[8 + i] = hi[i]; }
    return r;
}
static __device__ __forceinline__ float sigm(float x) {
    return 1.0f / (1.0f + __expf(-x));
}
static __device__ __forceinline__ float tanh_fast(float x) {
    return 1.0f - 2.0f / (__expf(2.0f * x) + 1.0f);
}

// ---------------------------------------------------------------------------
// TDM: async 2D tile load gx[row0..row0+15][0..767] (bf16) -> LDS
//   row stride in elements = T_*G_ ; tile = 768 x 16 ; data_size = 2 bytes
// ---------------------------------------------------------------------------
static __device__ __forceinline__ void tdm_load_gx_tile(
        const unsigned short* gptr,  // &gx[(m0*T + t)*G]
        unsigned lds_byte_off) {
    unsigned long long ga = (unsigned long long)(const void*)gptr;
    u32x4 g0;
    g0[0] = 1u;                                   // count=1 (valid descriptor)
    g0[1] = lds_byte_off;                         // lds_addr
    g0[2] = (unsigned)ga;                         // global_addr[31:0]
    g0[3] = (unsigned)((ga >> 32) & 0x1FFFFFFu)   // global_addr[56:32]
          | (2u << 30);                           // type=2 ("image")
    i32x8 g1;
    g1[0] = 0x00010000;              // workgroup_mask=0, data_size=1 (2 bytes)
    g1[1] = (int)(((unsigned)G_ & 0xFFFFu) << 16);        // tensor_dim0 lo16
    g1[2] = (int)(G_ >> 16);                              // tensor_dim0 hi16 (=0)
    g1[3] = (int)(((unsigned)(B_ * T_)) >> 16)            // tensor_dim1 hi16
          | (int)(((unsigned)G_ & 0xFFFFu) << 16);        // tile_dim0 = 768
    g1[4] = 16;                                           // tile_dim1=16, tile_dim2=0
    g1[5] = (int)(T_ * G_);                               // tensor_dim0_stride lo32
    g1[6] = 0;
    g1[7] = 0;
    i32x4 gz = (i32x4)(0);
#if __has_include(<hip/amd_detail/amd_gfx1250_TDM.h>)
    i32x8 gz8 = (i32x8)(0);
    __builtin_amdgcn_tensor_load_to_lds(g0, g1, gz, gz, gz8, 0);
#else
    __builtin_amdgcn_tensor_load_to_lds(g0, g1, gz, gz, 0);
#endif
}

// ---------------------------------------------------------------------------
// Kernel 0: convert weights fp32 -> bf16
// ---------------------------------------------------------------------------
__global__ void cvt_weights(const float* __restrict__ Wx,
                            const float* __restrict__ Wh,
                            const float* __restrict__ Wa,
                            unsigned short* __restrict__ Wxb,
                            unsigned short* __restrict__ Whb,
                            unsigned short* __restrict__ Wab) {
    const int n1 = G_ * D_;
    const int n2 = G_ * H_;
    const int n3 = H_ * 2 * H_;
    int stride = gridDim.x * blockDim.x;
    for (int i = blockIdx.x * blockDim.x + threadIdx.x; i < n1; i += stride)
        Wxb[i] = f2bfu(Wx[i]);
    for (int i = blockIdx.x * blockDim.x + threadIdx.x; i < n2; i += stride)
        Whb[i] = f2bfu(Wh[i]);
    for (int i = blockIdx.x * blockDim.x + threadIdx.x; i < n3; i += stride)
        Wab[i] = f2bfu(Wa[i]);
}

// ---------------------------------------------------------------------------
// Kernel 1: gx = x @ Wx^T + bx   -> bf16 [B*T, 768]
//   Each wave: 16 rows x 12 col-tiles. Block: 8 waves. grid = (4096, 4)
// ---------------------------------------------------------------------------
__global__ void __launch_bounds__(256) gx_gemm(
        const float* __restrict__ x,
        const unsigned short* __restrict__ Wxb,
        const float* __restrict__ bx,
        unsigned short* __restrict__ gx) {
    const int w    = threadIdx.x >> 5;
    const int lane = threadIdx.x & 31;
    const int lr   = lane & 15;
    const int lh   = lane >> 4;

    const int row0    = (blockIdx.x * 8 + w) * 16;
    const int colBase = blockIdx.y * 192;

    v8f acc[12];
#pragma unroll
    for (int i = 0; i < 12; ++i) acc[i] = (v8f)(0.0f);

    const float* xrow = x + (size_t)(row0 + lr) * D_;

#pragma unroll
    for (int kt = 0; kt < 8; ++kt) {
        const int k0 = kt * 32;
        const float* xp = xrow + k0 + lh * 8;
        v4f f0 = *(const v4f*)(xp);
        v4f f1 = *(const v4f*)(xp + 4);
        v4f f2 = *(const v4f*)(xp + 16);
        v4f f3 = *(const v4f*)(xp + 20);
        v16bf a;
#pragma unroll
        for (int e = 0; e < 4; ++e) {
            a[e]      = (__bf16)f0[e];
            a[4 + e]  = (__bf16)f1[e];
            a[8 + e]  = (__bf16)f2[e];
            a[12 + e] = (__bf16)f3[e];
        }
#pragma unroll
        for (int i = 0; i < 12; ++i) {
            const int j = colBase + i * 16 + lr;
            const unsigned short* pb = Wxb + (size_t)j * D_ + k0 + lh * 8;
            v16bf b = mk16(*(const v8bf*)pb, *(const v8bf*)(pb + 16));
            acc[i] = __builtin_amdgcn_wmma_f32_16x16x32_bf16(
                false, a, false, b, (short)0, acc[i], false, false);
        }
    }

#pragma unroll
    for (int i = 0; i < 12; ++i) {
        const int j  = colBase + i * 16 + lr;
        const float bias = bx[j];
#pragma unroll
        for (int r = 0; r < 8; ++r) {
            const int rr = row0 + lh * 8 + r;
            gx[(size_t)rr * G_ + j] = f2bfu(acc[i][r] + bias);
        }
    }
}

// ---------------------------------------------------------------------------
// Kernel 2: fused 512-step scan, TDM double-buffered gx tiles.
//   grid = 64 blocks x 256 threads (8 waves); block owns 16 batch rows.
// ---------------------------------------------------------------------------
__global__ void __launch_bounds__(256) gru_scan(
        const unsigned short* __restrict__ gx,   // [B*T, 768] bf16
        const unsigned short* __restrict__ Whb,  // [768, 256] bf16
        const float* __restrict__ bh,
        const unsigned short* __restrict__ Wab,  // [256, 512] bf16
        const float* __restrict__ ba,
        float* __restrict__ out) {               // [B, T, 256]
    __shared__ unsigned short gxS[2][16][G_];  // 48 KB  (TDM double buffer)
    __shared__ float          ghS[16][2 * H_]; // 32 KB  (h_r | h_n, fp32)
    __shared__ unsigned short ghiS[16][H_];    //  8 KB  (h_i, bf16 att A-frags)
    __shared__ float          attS[16][H_];    // 16 KB
    __shared__ float          hF[16][H_];      // 16 KB
    __shared__ unsigned short hB[16][H_];      //  8 KB   -> total 128 KB

    const int w    = threadIdx.x >> 5;
    const int lane = threadIdx.x & 31;
    const int lr   = lane & 15;
    const int lh   = lane >> 4;
    const int m0   = blockIdx.x * 16;

    float bhv[6];
#pragma unroll
    for (int i = 0; i < 6; ++i) bhv[i] = bh[(6 * w + i) * 16 + lr];
    float bav[2];
#pragma unroll
    for (int i = 0; i < 2; ++i) bav[i] = ba[(2 * w + i) * 16 + lr];

    for (int idx = threadIdx.x; idx < 16 * H_; idx += 256) {
        const int r = idx >> 8, c = idx & 255;
        hF[r][c] = 0.0f;
        hB[r][c] = 0;
    }

    // Kick off TDM for t=0 (wave 0 issues; TDM ignores EXEC)
    if (w == 0) {
        tdm_load_gx_tile(gx + ((size_t)m0 * T_ + 0) * G_,
                         (unsigned)(size_t)&gxS[0][0][0]);
    }
    __syncthreads();

    for (int t = 0; t < T_; ++t) {
        const int sel = t & 1;
        // Wait for this step's gx tile, then prefetch next step's tile.
        if (w == 0) {
            __builtin_amdgcn_s_wait_tensorcnt(0);
            if (t + 1 < T_)
                tdm_load_gx_tile(gx + ((size_t)m0 * T_ + (t + 1)) * G_,
                                 (unsigned)(size_t)&gxS[sel ^ 1][0][0]);
        }
        __syncthreads();

        // ---- gh = h @ Wh^T + bh ------------------------------------------
        v8f acc[6];
#pragma unroll
        for (int i = 0; i < 6; ++i) acc[i] = (v8f)(0.0f);
#pragma unroll
        for (int kt = 0; kt < 8; ++kt) {
            const int k0 = kt * 32;
            const unsigned short* pa = &hB[lr][k0 + lh * 8];
            v16bf a = mk16(*(const v8bf*)pa, *(const v8bf*)(pa + 16));
#pragma unroll
            for (int i = 0; i < 6; ++i) {
                const int j = (6 * w + i) * 16 + lr;
                const unsigned short* pb = Whb + (size_t)j * H_ + k0 + lh * 8;
                v16bf b = mk16(*(const v8bf*)pb, *(const v8bf*)(pb + 16));
                acc[i] = __builtin_amdgcn_wmma_f32_16x16x32_bf16(
                    false, a, false, b, (short)0, acc[i], false, false);
            }
        }
#pragma unroll
        for (int i = 0; i < 6; ++i) {
            const int jc = (6 * w + i) * 16 + lr;
#pragma unroll
            for (int r = 0; r < 8; ++r) {
                const int rr = lh * 8 + r;
                const float v = acc[i][r] + bhv[i];
                if (jc < H_)            ghS[rr][jc] = v;           // h_r
                else if (jc < 2 * H_)   ghiS[rr][jc - H_] = f2bfu(v); // h_i
                else                    ghS[rr][jc - H_] = v;      // h_n at +256
            }
        }
        __syncthreads();

        // ---- att = sigmoid(concat(i_i, h_i) @ Wa^T + ba) ------------------
        v8f acc2[2];
#pragma unroll
        for (int i = 0; i < 2; ++i) acc2[i] = (v8f)(0.0f);
#pragma unroll
        for (int kt = 0; kt < 16; ++kt) {
            const int k0 = kt * 32;
            v16bf a;
            if (kt < 8) {
                const unsigned short* pa = &gxS[sel][lr][H_ + k0 + lh * 8];
                a = mk16(*(const v8bf*)pa, *(const v8bf*)(pa + 16));
            } else {
                const int kl = (kt - 8) * 32 + lh * 8;
                const unsigned short* pa = &ghiS[lr][kl];
                a = mk16(*(const v8bf*)pa, *(const v8bf*)(pa + 16));
            }
#pragma unroll
            for (int i = 0; i < 2; ++i) {
                const int j = (2 * w + i) * 16 + lr;
                const unsigned short* pb = Wab + (size_t)j * (2 * H_) + k0 + lh * 8;
                v16bf b = mk16(*(const v8bf*)pb, *(const v8bf*)(pb + 16));
                acc2[i] = __builtin_amdgcn_wmma_f32_16x16x32_bf16(
                    false, a, false, b, (short)0, acc2[i], false, false);
            }
        }
#pragma unroll
        for (int i = 0; i < 2; ++i) {
            const int jc = (2 * w + i) * 16 + lr;
#pragma unroll
            for (int r = 0; r < 8; ++r)
                attS[lh * 8 + r][jc] = sigm(acc2[i][r] + bav[i]);
        }
        __syncthreads();

        // ---- elementwise GRU update (all gate inputs now in LDS) ----------
#pragma unroll 4
        for (int rep = 0; rep < 16; ++rep) {
            const int idx = threadIdx.x + rep * 256;
            const int r = idx >> 8, c = idx & 255;
            const int brow = m0 + r;
            const float i_r = bf2f(gxS[sel][r][c]);
            const float i_n = bf2f(gxS[sel][r][2 * H_ + c]);
            const float h_r = ghS[r][c];
            const float h_n = ghS[r][H_ + c];
            const float rg  = sigm(i_r + h_r);
            const float ng  = tanh_fast(i_n + rg * h_n);
            const float ig  = attS[r][c];
            const float ho  = hF[r][c];
            const float hy  = ng + ig * (ho - ng);
            out[((size_t)brow * T_ + t) * H_ + c] = hy;
            hF[r][c] = hy;
            hB[r][c] = f2bfu(hy);
        }
        __syncthreads();
    }
}

// ---------------------------------------------------------------------------
// Host launcher
// ---------------------------------------------------------------------------
extern "C" void kernel_launch(void* const* d_in, const int* in_sizes, int n_in,
                              void* d_out, int out_size, void* d_ws, size_t ws_size,
                              hipStream_t stream) {
    const float* x  = (const float*)d_in[0];
    const float* Wx = (const float*)d_in[1];
    const float* bx = (const float*)d_in[2];
    const float* Wh = (const float*)d_in[3];
    const float* bh = (const float*)d_in[4];
    const float* Wa = (const float*)d_in[5];
    const float* ba = (const float*)d_in[6];
    float* out = (float*)d_out;

    char* ws = (char*)d_ws;
    unsigned short* gxb = (unsigned short*)ws;                     // 768 MB
    size_t off = (size_t)B_ * T_ * G_ * sizeof(unsigned short);
    unsigned short* Wxb = (unsigned short*)(ws + off); off += (size_t)G_ * D_ * 2;
    unsigned short* Whb = (unsigned short*)(ws + off); off += (size_t)G_ * H_ * 2;
    unsigned short* Wab = (unsigned short*)(ws + off);

    cvt_weights<<<512, 256, 0, stream>>>(Wx, Wh, Wa, Wxb, Whb, Wab);
    gx_gemm<<<dim3(4096, 4), 256, 0, stream>>>(x, Wxb, bx, gxb);
    gru_scan<<<64, 256, 0, stream>>>(gxb, Whb, bh, Wab, ba, out);
}
